// GraphEncoder_GAT_824633721631
// MI455X (gfx1250) — matrix operand order repaired and checked by reading, not compiled, not run on hardware
//
#include <hip/hip_runtime.h>
#include <math.h>

// ---------------------------------------------------------------------------
// GAT 2-layer pipeline for MI455X (gfx1250, wave32).
// GEMMs -> v_wmma_f32_16x16x32_bf16 fed by b128 loads of pre-packed bf16
// operands (A row-major bf16, B transposed bf16), fp32 accumulation.
// Sparse softmax/aggregation -> fp32 atomics (HBM/atomic bound, ~205M adds).
// ---------------------------------------------------------------------------

typedef __attribute__((ext_vector_type(16))) __bf16 v16bf;
typedef __attribute__((ext_vector_type(8)))  __bf16 v8bf;
typedef __attribute__((ext_vector_type(8)))  float  v8f;

#define LRELU_SLOPE 0.2f

// ------------------------------- utility -----------------------------------
__global__ void fill_kernel(float* __restrict__ p, float v, int n) {
  int i = blockIdx.x * blockDim.x + threadIdx.x;
  if (i < n) p[i] = v;
}

// fp32 -> bf16, 8 elements per thread (n % 8 == 0)
__global__ void cvt_bf16_kernel(const float* __restrict__ in,
                                __bf16* __restrict__ out, int n) {
  int i8 = (blockIdx.x * blockDim.x + threadIdx.x) * 8;
  if (i8 >= n) return;
  float4 a = *(const float4*)(in + i8);
  float4 b = *(const float4*)(in + i8 + 4);
  v8bf o;
  o[0] = (__bf16)a.x; o[1] = (__bf16)a.y; o[2] = (__bf16)a.z; o[3] = (__bf16)a.w;
  o[4] = (__bf16)b.x; o[5] = (__bf16)b.y; o[6] = (__bf16)b.z; o[7] = (__bf16)b.w;
  *(v8bf*)(out + i8) = o;
}

// W (K x M, fp32) -> Wt (M x K, bf16)
__global__ void transpose_cvt_kernel(const float* __restrict__ W,
                                     __bf16* __restrict__ Wt, int K, int M) {
  int i = blockIdx.x * blockDim.x + threadIdx.x;
  if (i >= K * M) return;
  int m = i / K, k = i % K;
  Wt[i] = (__bf16)W[(size_t)k * M + m];
}

// ------------------------------- GEMM (WMMA) --------------------------------
// C[N x M] = A[N x K] * Bt[M x K]^T ; A,Bt bf16, C fp32.
// blockDim = 128 (4 waves); each wave owns 16 rows x 64 cols (4 accumulators
// sharing one A fragment). grid = (ceil(N/64), M/64). K%32==0, M%64==0.
__global__ void gemm_bf16_wmma(const __bf16* __restrict__ A,
                               const __bf16* __restrict__ Bt,
                               float* __restrict__ C,
                               int N, int K, int M) {
  const int lane = threadIdx.x & 31;
  const int wave = threadIdx.x >> 5;
  const int row0 = (blockIdx.x * 4 + wave) * 16;
  if (row0 >= N) return;                 // wave-uniform: EXEC stays all-1s
  const int col0 = blockIdx.y * 64;
  const int half = lane >> 4;            // 0: lanes 0-15, 1: lanes 16-31
  const int lrow = lane & 15;

  v8f zero = {};
  v8f acc[4];
  acc[0] = zero; acc[1] = zero; acc[2] = zero; acc[3] = zero;

  int m = row0 + lrow;
  if (m >= N) m = N - 1;                 // clamp loads; stores are guarded
  const __bf16* __restrict__ Arow = A + (size_t)m * K;
  const __bf16* __restrict__ Bn0  = Bt + (size_t)(col0 + lrow) * K;

  for (int kk = 0; kk < K; kk += 32) {
    // A fragment (16-bit A 16x32, ISA 7.12.2): two contiguous 8-elem K runs:
    //   j in 0..7  -> k = kk + 8*half + j
    //   j in 8..15 -> k = kk + 16 + 8*half + (j-8)
    v8bf alo = *(const v8bf*)(Arow + kk + 8 * half);
    v8bf ahi = *(const v8bf*)(Arow + kk + 16 + 8 * half);
    v16bf afrag = __builtin_shufflevector(alo, ahi,
        0, 1, 2, 3, 4, 5, 6, 7, 8, 9, 10, 11, 12, 13, 14, 15);

    // B: same K order as A: j<8 -> kk+8*half+j; j>=8 -> kk+16+8*half+(j-8)
#pragma unroll
    for (int t = 0; t < 4; ++t) {
      const __bf16* Brow = Bn0 + (size_t)t * 16 * K;   // column n = col0+t*16+lrow
      v8bf blo = *(const v8bf*)(Brow + kk + 8 * half);
      v8bf bhi = *(const v8bf*)(Brow + kk + 16 + 8 * half);
      v16bf bfrag = __builtin_shufflevector(blo, bhi,
          0, 1, 2, 3, 4, 5, 6, 7, 8, 9, 10, 11, 12, 13, 14, 15);
      acc[t] = __builtin_amdgcn_wmma_f32_16x16x32_bf16(
          /*neg_a=*/false, afrag, /*neg_b=*/false, bfrag,
          /*c_mod=*/(short)0, acc[t], /*reuse_a=*/false, /*reuse_b=*/false);
    }
  }

  // C/D layout: VGPR r -> row = r + 8*half; col = lane%16
#pragma unroll
  for (int t = 0; t < 4; ++t) {
    int n = col0 + t * 16 + lrow;
#pragma unroll
    for (int r = 0; r < 8; ++r) {
      int row = row0 + r + 8 * half;
      if (row < N) C[(size_t)row * M + n] = acc[t][r];
    }
  }
}

// --------------------------- attention dot products -------------------------
__global__ void attn_dot_kernel(const float* __restrict__ feat,
                                const float* __restrict__ attn_l,
                                const float* __restrict__ attn_r,
                                float* __restrict__ el, float* __restrict__ er,
                                int N, int H, int D) {
  int i = blockIdx.x * blockDim.x + threadIdx.x;
  if (i >= N * H) return;
  int n = i / H, h = i % H;
  const float* f  = feat + (size_t)n * H * D + (size_t)h * D;
  const float* al = attn_l + h * D;
  const float* ar = attn_r + h * D;
  float sl = 0.f, sr = 0.f;
  for (int d = 0; d < D; d += 4) {
    float4 fv = *(const float4*)(f + d);
    float4 av = *(const float4*)(al + d);
    float4 rv = *(const float4*)(ar + d);
    sl += fv.x * av.x + fv.y * av.y + fv.z * av.z + fv.w * av.w;
    sr += fv.x * rv.x + fv.y * rv.y + fv.z * rv.z + fv.w * rv.w;
  }
  el[i] = sl; er[i] = sr;
}

// --------------------------- edge scores + segment max -----------------------
__device__ __forceinline__ void atomic_max_f32(float* addr, float v) {
  // sign-split trick: valid given init to -inf
  if (v >= 0.f) atomicMax((int*)addr, __float_as_int(v));
  else          atomicMin((unsigned int*)addr, __float_as_uint(v));
}

__global__ void edge_score_kernel(const int* __restrict__ src,
                                  const int* __restrict__ dst,
                                  const float* __restrict__ el,
                                  const float* __restrict__ er,
                                  float* __restrict__ eval,   // out: leaky-relu logit
                                  float* __restrict__ mmax,   // init -inf
                                  int E, int H) {
  int i = blockIdx.x * blockDim.x + threadIdx.x;
  if (i >= E * H) return;
  int e = i / H, h = i % H;
  int sn = src[e], dn = dst[e];
  float v = el[sn * H + h] + er[dn * H + h];
  v = (v > 0.f) ? v : LRELU_SLOPE * v;
  eval[i] = v;
  atomic_max_f32(mmax + dn * H + h, v);
}

// --------------------------- exp + segment sum -------------------------------
__global__ void edge_exp_kernel(const int* __restrict__ dst,
                                const float* __restrict__ mmax,
                                float* __restrict__ eval,     // in: logit, out: exp
                                float* __restrict__ ssum,     // init 0
                                int E, int H) {
  int i = blockIdx.x * blockDim.x + threadIdx.x;
  if (i >= E * H) return;
  int e = i / H, h = i % H;
  int dn = dst[e];
  float ex = expf(eval[i] - mmax[dn * H + h]);
  eval[i] = ex;
  atomicAdd(ssum + dn * H + h, ex);
}

// --------------------------- weighted message aggregation --------------------
// One wave per edge; lane owns a contiguous (H*D/32)-float slice.
__global__ void aggregate_kernel(const int* __restrict__ src,
                                 const int* __restrict__ dst,
                                 const float* __restrict__ ex,
                                 const float* __restrict__ ssum,
                                 const float* __restrict__ feat,
                                 float* __restrict__ rst,
                                 int E, int H, int D) {
  int wavesPerBlock = blockDim.x >> 5;
  int wave = threadIdx.x >> 5;
  int lane = threadIdx.x & 31;
  int e = blockIdx.x * wavesPerBlock + wave;
  if (e >= E) return;
  int sn = src[e], dn = dst[e];
  int FD = H * D;
  int perLane = FD >> 5;                 // 8 (layer1) or 4 (layer2)
  const float* fs = feat + (size_t)sn * FD;
  float*       rd = rst  + (size_t)dn * FD;
  for (int c = 0; c < perLane; c += 4) {
    int d0 = lane * perLane + c;
    int h  = d0 / D;                     // 4-aligned chunk never crosses a head
    float alpha = ex[(size_t)e * H + h] / ssum[dn * H + h];
    float4 fv = *(const float4*)(fs + d0);
    atomicAdd(rd + d0 + 0, fv.x * alpha);
    atomicAdd(rd + d0 + 1, fv.y * alpha);
    atomicAdd(rd + d0 + 2, fv.z * alpha);
    atomicAdd(rd + d0 + 3, fv.w * alpha);
  }
}

// ------------------ bias + relu, fp32 in -> bf16 out (layer 1) ---------------
__global__ void bias_relu_cvt_kernel(const float* __restrict__ x,
                                     const float* __restrict__ bias,
                                     __bf16* __restrict__ out,
                                     int N, int FD) {
  int i = blockIdx.x * blockDim.x + threadIdx.x;
  if (i >= N * FD) return;
  float v = x[i] + bias[i % FD];
  out[i] = (__bf16)fmaxf(v, 0.f);
}

// --------------------------- bias add, in place (layer 2) --------------------
__global__ void bias_kernel(float* __restrict__ x, const float* __restrict__ bias,
                            int N, int FD) {
  int i = blockIdx.x * blockDim.x + threadIdx.x;
  if (i >= N * FD) return;
  x[i] = x[i] + bias[i % FD];
}

// ---------------------------------------------------------------------------
extern "C" void kernel_launch(void* const* d_in, const int* in_sizes, int n_in,
                              void* d_out, int out_size, void* d_ws, size_t ws_size,
                              hipStream_t stream) {
  const float* node_emb = (const float*)d_in[0];
  const int*   src      = (const int*)d_in[1];
  const int*   dst      = (const int*)d_in[2];
  const float* W1       = (const float*)d_in[3];
  const float* attn_l1  = (const float*)d_in[4];
  const float* attn_r1  = (const float*)d_in[5];
  const float* bias1    = (const float*)d_in[6];
  const float* W2       = (const float*)d_in[7];
  const float* attn_l2  = (const float*)d_in[8];
  const float* attn_r2  = (const float*)d_in[9];
  const float* bias2    = (const float*)d_in[10];
  float* out = (float*)d_out;

  const int IN = 128, H1 = 4, D1 = 64, F1 = 256, H2 = 1, D2 = 128, F2 = 128;
  const int N = in_sizes[0] / IN;
  const int E = in_sizes[1];

  char* ws = (char*)d_ws;
  size_t off = 0;
  auto alloc = [&](size_t bytes) -> void* {
    void* p = (void*)(ws + off);
    off += (bytes + 255) & ~(size_t)255;
    return p;
  };
  float*  feat1 = (float*)alloc((size_t)N * F1 * 4);    // X @ W1 (fp32)
  float*  h1    = (float*)alloc((size_t)N * F1 * 4);    // layer1 aggregation target
  float*  feat2 = (float*)alloc((size_t)N * F2 * 4);    // h1 @ W2 (fp32)
  __bf16* Xbf   = (__bf16*)alloc((size_t)N * IN * 2);   // node_emb as bf16
  __bf16* h1bf  = (__bf16*)alloc((size_t)N * F1 * 2);   // relu(h1+bias) as bf16
  __bf16* W1t   = (__bf16*)alloc((size_t)F1 * IN * 2);  // W1^T bf16 (M x K)
  __bf16* W2t   = (__bf16*)alloc((size_t)F2 * F1 * 2);  // W2^T bf16 (M x K)
  float* el1 = (float*)alloc((size_t)N * H1 * 4);
  float* er1 = (float*)alloc((size_t)N * H1 * 4);
  float* m1  = (float*)alloc((size_t)N * H1 * 4);
  float* s1  = (float*)alloc((size_t)N * H1 * 4);
  float* el2 = (float*)alloc((size_t)N * 4);
  float* er2 = (float*)alloc((size_t)N * 4);
  float* m2  = (float*)alloc((size_t)N * 4);
  float* s2  = (float*)alloc((size_t)N * 4);
  float* ex1 = (float*)alloc((size_t)E * H1 * 4);
  float* ex2 = (float*)alloc((size_t)E * 4);

  const float NEG_INF = -__builtin_inff();
  auto cdiv = [](int a, int b) { return (a + b - 1) / b; };

  // ---------------- operand packing ----------------
  cvt_bf16_kernel<<<cdiv(N * IN / 8, 256), 256, 0, stream>>>(node_emb, Xbf, N * IN);
  transpose_cvt_kernel<<<cdiv(IN * F1, 256), 256, 0, stream>>>(W1, W1t, IN, F1);
  transpose_cvt_kernel<<<cdiv(F1 * F2, 256), 256, 0, stream>>>(W2, W2t, F1, F2);

  // ---------------- layer 1 ----------------
  {
    dim3 g((unsigned)cdiv(N, 64), (unsigned)(F1 / 64));
    gemm_bf16_wmma<<<g, 128, 0, stream>>>(Xbf, W1t, feat1, N, IN, F1);
  }
  attn_dot_kernel<<<cdiv(N * H1, 256), 256, 0, stream>>>(feat1, attn_l1, attn_r1,
                                                         el1, er1, N, H1, D1);
  fill_kernel<<<cdiv(N * H1, 256), 256, 0, stream>>>(m1, NEG_INF, N * H1);
  fill_kernel<<<cdiv(N * H1, 256), 256, 0, stream>>>(s1, 0.f, N * H1);
  fill_kernel<<<cdiv(N * F1, 256), 256, 0, stream>>>(h1, 0.f, N * F1);
  edge_score_kernel<<<cdiv(E * H1, 256), 256, 0, stream>>>(src, dst, el1, er1,
                                                           ex1, m1, E, H1);
  edge_exp_kernel<<<cdiv(E * H1, 256), 256, 0, stream>>>(dst, m1, ex1, s1, E, H1);
  aggregate_kernel<<<cdiv(E, 8), 256, 0, stream>>>(src, dst, ex1, s1, feat1, h1,
                                                   E, H1, D1);
  bias_relu_cvt_kernel<<<cdiv(N * F1, 256), 256, 0, stream>>>(h1, bias1, h1bf, N, F1);

  // ---------------- layer 2 ----------------
  {
    dim3 g((unsigned)cdiv(N, 64), (unsigned)(F2 / 64));
    gemm_bf16_wmma<<<g, 128, 0, stream>>>(h1bf, W2t, feat2, N, F1, F2);
  }
  attn_dot_kernel<<<cdiv(N * H2, 256), 256, 0, stream>>>(feat2, attn_l2, attn_r2,
                                                         el2, er2, N, H2, D2);
  fill_kernel<<<cdiv(N, 256), 256, 0, stream>>>(m2, NEG_INF, N);
  fill_kernel<<<cdiv(N, 256), 256, 0, stream>>>(s2, 0.f, N);
  fill_kernel<<<cdiv(N * F2, 256), 256, 0, stream>>>(out, 0.f, N * F2);
  edge_score_kernel<<<cdiv(E, 256), 256, 0, stream>>>(src, dst, el2, er2,
                                                      ex2, m2, E, H2);
  edge_exp_kernel<<<cdiv(E, 256), 256, 0, stream>>>(dst, m2, ex2, s2, E, H2);
  aggregate_kernel<<<cdiv(E, 8), 256, 0, stream>>>(src, dst, ex2, s2, feat2, out,
                                                   E, H2, D2);
  // mean over 1 head == identity; just add bias2
  bias_kernel<<<cdiv(N * F2, 256), 256, 0, stream>>>(out, bias2, N, F2);
}
